// Encoder_Model_15874199126585
// MI455X (gfx1250) — matrix operand order repaired and checked by reading, not compiled, not run on hardware
//
#include <hip/hip_runtime.h>
#include <hip/hip_bf16.h>
#include <math.h>

// ---------------------------------------------------------------------------
// Problem constants (match the reference)
// ---------------------------------------------------------------------------
constexpr int kNodes = 100000;
constexpr int kDim   = 512;
constexpr int kPairs = 2048;
constexpr int kRows  = 2 * kPairs;          // rows 0..2047 = l-side, 2048..4095 = r-side
constexpr float kGamma = 3.0f;
constexpr float kLamb  = 20.0f;
constexpr float kTau   = 8.0f;

constexpr int kNTiles  = kNodes / 16;       // 6250 node tiles (exact)
constexpr int kNChunk  = 5;                 // node chunks -> 41 MB bf16(hi+lo) each, L2-resident
constexpr int kTPC     = kNTiles / kNChunk; // 1250 tiles per chunk (exact)
constexpr int kPTiles  = kRows / 16;        // 256 pair tiles

// ---------------------------------------------------------------------------
// WMMA vector types (CDNA5 wave32)
// ---------------------------------------------------------------------------
typedef __attribute__((ext_vector_type(16))) __bf16 v16bf;
typedef __attribute__((ext_vector_type(8)))  __bf16 v8bf;
typedef __attribute__((ext_vector_type(8)))  float  v8f;

union V16 { v16bf v; v8bf h[2]; };

// ---------------------------------------------------------------------------
// K1: convert emb (f32) -> bf16 hi/lo split, and per-node squared norms.
// One wave per node row (512 elems = 16 per lane).
// ---------------------------------------------------------------------------
__global__ __launch_bounds__(256)
void k_convert(const float* __restrict__ emb, __bf16* __restrict__ bhi,
               __bf16* __restrict__ blo, float* __restrict__ embsq) {
  const int wv = threadIdx.x >> 5, lane = threadIdx.x & 31;
  const int node = blockIdx.x * 8 + wv;
  const float* r = emb + (size_t)node * kDim;
  __bf16* hi = bhi + (size_t)node * kDim;
  __bf16* lo = blo + (size_t)node * kDim;
  float sq = 0.f;
#pragma unroll
  for (int i = 0; i < kDim / 32; ++i) {
    const int k = lane + i * 32;
    const float x = r[k];
    sq += x * x;
    const __bf16 h = (__bf16)x;          // hi part (RNE)
    const float  res = x - (float)h;     // exact residual
    hi[k] = h;
    lo[k] = (__bf16)res;                 // lo part -> ~2^-16 rel error total
  }
#pragma unroll
  for (int off = 16; off > 0; off >>= 1) sq += __shfl_xor(sq, off, 32);
  if (lane == 0) embsq[node] = sq;
}

// ---------------------------------------------------------------------------
// K2: per-pair constants.  pos = ||e_l - e_r||^2,
//     crow[p]       = pos - ||e_l||^2 + gamma   (l-side row)
//     crow[p+2048]  = pos - ||e_r||^2 + gamma   (r-side row)
// One wave per pair.
// ---------------------------------------------------------------------------
__global__ __launch_bounds__(256)
void k_crow(const int* __restrict__ pairs, const float* __restrict__ emb,
            const float* __restrict__ embsq, float* __restrict__ crow) {
  const int wv = threadIdx.x >> 5, lane = threadIdx.x & 31;
  const int p = blockIdx.x * 8 + wv;
  const int l = pairs[2 * p], r = pairs[2 * p + 1];
  const float* el = emb + (size_t)l * kDim;
  const float* er = emb + (size_t)r * kDim;
  float pos = 0.f;
#pragma unroll
  for (int i = 0; i < kDim / 32; ++i) {
    const int k = lane + i * 32;
    const float d = el[k] - er[k];
    pos += d * d;
  }
#pragma unroll
  for (int off = 16; off > 0; off >>= 1) pos += __shfl_xor(pos, off, 32);
  if (lane == 0) {
    crow[p]          = pos - embsq[l] + kGamma;
    crow[p + kPairs] = pos - embsq[r] + kGamma;
  }
}

// ---------------------------------------------------------------------------
// Fused GEMM + statistics / LSE pass.
//   score(row, node) = mask * (crow[row] - ||e_node||^2 + 2 * <A_row, e_node>)
// A-operand (M) = 16 nodes, B-operand (N) = 16 pairs (staged once in LDS).
// Each lane owns one pair column -> scalar running accumulators.
// PASS 1: per-row sum / sumsq / max.   PASS 2: per-row sum exp(a*v + b).
// Partial results per (row, chunk) in global scratch.
// ---------------------------------------------------------------------------
template <int PASS>
__global__ __launch_bounds__(256)
void k_gemm(const __bf16* __restrict__ bhi, const __bf16* __restrict__ blo,
            const int* __restrict__ pairs, const float* __restrict__ embsq,
            const float* __restrict__ crow, const float* __restrict__ aC,
            const float* __restrict__ bC, float* __restrict__ o0,
            float* __restrict__ o1, float* __restrict__ o2) {
  __shared__ alignas(32) __bf16 sHi[16 * kDim];   // pair tile, hi parts (16 KB)
  __shared__ alignas(32) __bf16 sLo[16 * kDim];   // pair tile, lo parts (16 KB)
  __shared__ float redA[8][16], redB[8][16], redC[8][16];

  const int ptile = blockIdx.x;
  const int chunk = blockIdx.y;
  const int tid = threadIdx.x, lane = tid & 31, wv = tid >> 5;

  // Stage the 16 gathered pair rows (bf16 hi/lo) for this pair tile.
  for (int i = tid; i < 16 * kDim; i += 256) {
    const int rr = i >> 9, k = i & (kDim - 1);
    const int row = ptile * 16 + rr, pp = row & (kPairs - 1);
    const int node = (row < kPairs) ? pairs[2 * pp] : pairs[2 * pp + 1];
    sHi[i] = bhi[(size_t)node * kDim + k];
    sLo[i] = blo[(size_t)node * kDim + k];
  }
  __syncthreads();

  const int col = lane & 15;            // pair column owned by this lane
  const int kh  = (lane >> 4) & 1;      // half-wave selector
  const int row = ptile * 16 + col;
  const int pp  = row & (kPairs - 1);
  const int lv = pairs[2 * pp], rv = pairs[2 * pp + 1];
  const float cr = crow[row];
  float ea = 0.f, eb = 0.f;
  if (PASS == 2) { ea = aC[row]; eb = bC[row]; }

  float sm = 0.f, s2 = 0.f, mx = -3.402823466e38f, se = 0.f;

  const __bf16* lbh = sHi + col * kDim + kh * 16;  // B operand: 16 contig K per lane
  const __bf16* lbl = sLo + col * kDim + kh * 16;

  for (int t = chunk * kTPC + wv; t < (chunk + 1) * kTPC; t += 8) {
    const int n0 = t * 16;
    if (n0 + 128 < kNodes) {           // prefetch this wave's next node tile
      __builtin_prefetch(bhi + (size_t)(n0 + 128) * kDim, 0, 1);
      __builtin_prefetch(blo + (size_t)(n0 + 128) * kDim, 0, 1);
    }
    v8f acc = {0.f, 0.f, 0.f, 0.f, 0.f, 0.f, 0.f, 0.f};
    // A operand: 16 contig K per lane (lane<16 -> K [0..15]; lane>=16 -> [16..31]), matching B
    const __bf16* gahi = bhi + (size_t)(n0 + col) * kDim + kh * 16;
    const __bf16* galo = blo + (size_t)(n0 + col) * kDim + kh * 16;
#pragma unroll 4
    for (int kk = 0; kk < kDim; kk += 32) {
      V16 ah, al, bh, bl;
      ah.h[0] = *(const v8bf*)(gahi + kk);
      ah.h[1] = *(const v8bf*)(gahi + kk + 8);
      al.h[0] = *(const v8bf*)(galo + kk);
      al.h[1] = *(const v8bf*)(galo + kk + 8);
      bh.v = *(const v16bf*)(lbh + kk);
      bl.v = *(const v16bf*)(lbl + kk);
      // fp32-accurate bf16 split: hi*hi + hi*lo + lo*hi
      acc = __builtin_amdgcn_wmma_f32_16x16x32_bf16(false, ah.v, false, bh.v,
                                                    (short)0, acc, false, false);
      acc = __builtin_amdgcn_wmma_f32_16x16x32_bf16(false, ah.v, false, bl.v,
                                                    (short)0, acc, false, false);
      acc = __builtin_amdgcn_wmma_f32_16x16x32_bf16(false, al.v, false, bh.v,
                                                    (short)0, acc, false, false);
    }
    const v8f esq = *(const v8f*)(embsq + n0 + kh * 8);
#pragma unroll
    for (int i = 0; i < 8; ++i) {
      const int n = n0 + kh * 8 + i;                 // D layout: M = vgpr (+8 upper half)
      float v = 2.f * acc[i] + cr - esq[i];
      const float mfac = 1.f - (n == lv ? 1.f : 0.f) - (n == rv ? 1.f : 0.f);
      v *= mfac;                                      // in-stream mask (handles l==r -> -v)
      if (PASS == 1) { sm += v; s2 += v * v; mx = fmaxf(mx, v); }
      else           { se += __expf(fmaf(v, ea, eb)); }  // exp(lamb*(v - max)/sd) <= 1
    }
  }

  // Fold the two half-wave partials for the same pair, then across the 8 waves.
  if (PASS == 1) {
    sm += __shfl_xor(sm, 16, 32);
    s2 += __shfl_xor(s2, 16, 32);
    mx = fmaxf(mx, __shfl_xor(mx, 16, 32));
    if (lane < 16) { redA[wv][lane] = sm; redB[wv][lane] = s2; redC[wv][lane] = mx; }
  } else {
    se += __shfl_xor(se, 16, 32);
    if (lane < 16) redA[wv][lane] = se;
  }
  __syncthreads();
  if (tid < 16) {
    const int grow = ptile * 16 + tid;
    if (PASS == 1) {
      float S = 0.f, Q = 0.f, M = -3.402823466e38f;
      for (int w = 0; w < 8; ++w) {
        S += redA[w][tid]; Q += redB[w][tid]; M = fmaxf(M, redC[w][tid]);
      }
      o0[grow * kNChunk + chunk] = S;
      o1[grow * kNChunk + chunk] = Q;
      o2[grow * kNChunk + chunk] = M;
    } else {
      float S = 0.f;
      for (int w = 0; w < 8; ++w) S += redA[w][tid];
      o0[grow * kNChunk + chunk] = S;
    }
  }
}

// ---------------------------------------------------------------------------
// K4: merge per-chunk stats -> per-row exp coefficients.
//   mu = S/N, sd = sqrt(Q/N - mu^2)  (population std, ddof=0 like jnp.std)
//   pass2 term: exp(lamb*(v - max)/sd) = exp(aC*v + bC)
//   final row loss = Mz + tau + log(sum)
// ---------------------------------------------------------------------------
__global__ __launch_bounds__(256)
void k_stats(const float* __restrict__ p1s, const float* __restrict__ p1q,
             const float* __restrict__ p1m, float* __restrict__ aC,
             float* __restrict__ bC, float* __restrict__ Mz) {
  const int row = blockIdx.x * 256 + threadIdx.x;
  if (row >= kRows) return;
  float S = 0.f, Q = 0.f, M = -3.402823466e38f;
  for (int c = 0; c < kNChunk; ++c) {
    S += p1s[row * kNChunk + c];
    Q += p1q[row * kNChunk + c];
    M = fmaxf(M, p1m[row * kNChunk + c]);
  }
  const float inv = 1.0f / (float)kNodes;
  const float mu = S * inv;
  const float var = fmaxf(Q * inv - mu * mu, 1e-20f);
  const float isd = 1.0f / sqrtf(var);
  aC[row] = kLamb * isd;
  bC[row] = -kLamb * M * isd;
  Mz[row] = kLamb * (M - mu) * isd;
}

// ---------------------------------------------------------------------------
// K6: final scalar = mean over pairs of (l_loss + r_loss)
//              = (1/2048) * sum over all 4096 rows of (Mz + tau + log(sum_exp))
// ---------------------------------------------------------------------------
__global__ __launch_bounds__(256)
void k_final(const float* __restrict__ p2s, const float* __restrict__ Mz,
             float* __restrict__ out) {
  __shared__ float red[256];
  float acc = 0.f;
  for (int row = (int)threadIdx.x; row < kRows; row += 256) {
    float S = 0.f;
    for (int c = 0; c < kNChunk; ++c) S += p2s[row * kNChunk + c];
    acc += Mz[row] + kTau + logf(S);
  }
  red[threadIdx.x] = acc;
  __syncthreads();
  for (int s = 128; s > 0; s >>= 1) {
    if ((int)threadIdx.x < s) red[threadIdx.x] += red[threadIdx.x + s];
    __syncthreads();
  }
  if (threadIdx.x == 0) out[0] = red[0] / (float)kPairs;
}

// ---------------------------------------------------------------------------
// Launcher
// ---------------------------------------------------------------------------
extern "C" void kernel_launch(void* const* d_in, const int* in_sizes, int n_in,
                              void* d_out, int out_size, void* d_ws, size_t ws_size,
                              hipStream_t stream) {
  const int*   pairs = (const int*)d_in[0];     // (2048, 2) int32
  const float* emb   = (const float*)d_in[1];   // (100000, 512) f32

  char* w = (char*)d_ws;
  size_t off = 0;
  auto carve = [&](size_t bytes) -> char* {
    char* p = w + off;
    off += (bytes + 255) & ~(size_t)255;
    return p;
  };
  __bf16* bhi  = (__bf16*)carve((size_t)kNodes * kDim * sizeof(__bf16)); // 102.4 MB
  __bf16* blo  = (__bf16*)carve((size_t)kNodes * kDim * sizeof(__bf16)); // 102.4 MB
  float* embsq = (float*)carve((size_t)kNodes * sizeof(float));
  float* crow  = (float*)carve((size_t)kRows * sizeof(float));
  float* p1s   = (float*)carve((size_t)kRows * kNChunk * sizeof(float));
  float* p1q   = (float*)carve((size_t)kRows * kNChunk * sizeof(float));
  float* p1m   = (float*)carve((size_t)kRows * kNChunk * sizeof(float));
  float* aC    = (float*)carve((size_t)kRows * sizeof(float));
  float* bC    = (float*)carve((size_t)kRows * sizeof(float));
  float* Mz    = (float*)carve((size_t)kRows * sizeof(float));
  float* p2s   = (float*)carve((size_t)kRows * kNChunk * sizeof(float));
  (void)in_sizes; (void)n_in; (void)out_size; (void)ws_size;

  k_convert<<<kNodes / 8, 256, 0, stream>>>(emb, bhi, blo, embsq);
  k_crow<<<kPairs / 8, 256, 0, stream>>>(pairs, emb, embsq, crow);

  dim3 grid(kPTiles, kNChunk);
  k_gemm<1><<<grid, 256, 0, stream>>>(bhi, blo, pairs, embsq, crow,
                                      nullptr, nullptr, p1s, p1q, p1m);
  k_stats<<<(kRows + 255) / 256, 256, 0, stream>>>(p1s, p1q, p1m, aC, bC, Mz);
  k_gemm<2><<<grid, 256, 0, stream>>>(bhi, blo, pairs, embsq, crow,
                                      aC, bC, p2s, nullptr, nullptr);
  k_final<<<1, 256, 0, stream>>>(p2s, Mz, (float*)d_out);
}
